// NeptuneDecoding_6614249636111
// MI455X (gfx1250) — matrix operand order repaired and checked by reading, not compiled, not run on hardware
//
#include <hip/hip_runtime.h>

typedef __attribute__((ext_vector_type(16))) _Float16 v16h;
typedef __attribute__((ext_vector_type(8)))  _Float16 v8h;
typedef __attribute__((ext_vector_type(8)))  float    v8f;

#define DHEAD   128
#define CCHUNK  128
#define NSPLITS 64

// ---------------------------------------------------------------------------
// Phase 1: one workgroup per (b*h, chunk). 8 waves x 32 lanes.
// Wave w computes scores for K rows [16w, 16w+16) via WMMA, then the block
// cooperatively does chunk max, exp-sum, and the p.V partial sum.
// ---------------------------------------------------------------------------
__global__ __launch_bounds__(256, 2)
void fd_partial(const _Float16* __restrict__ q,
                const _Float16* __restrict__ k,
                const _Float16* __restrict__ v,
                float* __restrict__ m_ws,
                float* __restrict__ l_ws,
                float* __restrict__ o_ws,
                int S)
{
    const int chunk = blockIdx.x % NSPLITS;
    const int bh    = blockIdx.x / NSPLITS;

    const int tid   = threadIdx.x;
    const int wave  = tid >> 5;
    const int lane  = tid & 31;
    const int lhalf = lane >> 4;   // 0: lower 16 lanes, 1: upper
    const int lrow  = lane & 15;

    const size_t kvoff = ((size_t)bh * (size_t)S + (size_t)chunk * CCHUNK) * DHEAD;
    const _Float16* kbase = k + kvoff;
    const _Float16* vbase = v + kvoff;
    const _Float16* qrow  = q + (size_t)bh * DHEAD;

    __shared__ float red_m[8];
    __shared__ float red_l[8];
    __shared__ float p_sh[CCHUNK];
    __shared__ float o_sh[DHEAD];

    // ---- QK^T via WMMA: wave handles rows c0..c0+15, accumulate over D in 4 steps of K=32
    const int c0 = wave * 16;
    const _Float16* krow = kbase + (size_t)(c0 + lrow) * DHEAD;

    v8f acc = {};
#pragma unroll
    for (int dk = 0; dk < 4; ++dk) {
        const int d0 = dk * 32 + lhalf * 8;
        // A-matrix (16x32 f16, ISA layout): two contiguous 16B segments per lane
        v8h alo = __builtin_nontemporal_load((const v8h*)(krow + d0));
        v8h ahi = __builtin_nontemporal_load((const v8h*)(krow + d0 + 16));
        v16h a;
#pragma unroll
        for (int i = 0; i < 8; ++i) { a[i] = alo[i]; a[i + 8] = ahi[i]; }
        // B-matrix (32x16 f16): every column n = q slice -> all D columns equal
        v8h blo = *(const v8h*)(qrow + d0);
        v8h bhi = *(const v8h*)(qrow + d0 + 16);
        v16h b;
#pragma unroll
        for (int i = 0; i < 8; ++i) { b[i] = blo[i]; b[i + 8] = bhi[i]; }
        acc = __builtin_amdgcn_wmma_f32_16x16x32_f16(
            /*neg_a=*/false, a, /*neg_b=*/false, b,
            /*c_mod=*/(short)0, acc, /*reuse_a=*/false, /*reuse_b=*/false);
    }

    // C layout: VGPR r, lanes 0-15 -> row c0+r ; lanes 16-31 -> row c0+8+r
    const float scale = 0.08838834764831845f;   // 1/sqrt(128)
    float s[8];
    float mw = -1e30f;
#pragma unroll
    for (int r = 0; r < 8; ++r) { s[r] = acc[r] * scale; mw = fmaxf(mw, s[r]); }
    mw = fmaxf(mw, __shfl_xor(mw, 16));         // combine the two 8-row halves

    if (lane == 0) red_m[wave] = mw;
    __syncthreads();

    float mchunk = red_m[0];
#pragma unroll
    for (int w2 = 1; w2 < 8; ++w2) mchunk = fmaxf(mchunk, red_m[w2]);

    // ---- p = exp(s - m), partial expsum
    float p[8];
    float lsum = 0.f;
#pragma unroll
    for (int r = 0; r < 8; ++r) { p[r] = __expf(s[r] - mchunk); lsum += p[r]; }
    lsum += __shfl_xor(lsum, 16);               // sum over this wave's 16 rows

    if (lane == 0) red_l[wave] = lsum;
    if (lrow == 0) {                            // one lane per half publishes its 8 rows
#pragma unroll
        for (int r = 0; r < 8; ++r) p_sh[c0 + lhalf * 8 + r] = p[r];
    }
    if (tid < DHEAD) o_sh[tid] = 0.f;
    __syncthreads();

    float lchunk = 0.f;
#pragma unroll
    for (int w2 = 0; w2 < 8; ++w2) lchunk += red_l[w2];

    // ---- o = sum_c p[c] * V[c][:]: thread t covers d-block (t&15), c-subset (t>>4)
    const int dblk = tid & 15;
    const int csub = tid >> 4;
    const int d0   = dblk * 8;
    const _Float16* vrow = vbase + (size_t)(csub * 8) * DHEAD + d0;

    float oacc[8] = {0.f, 0.f, 0.f, 0.f, 0.f, 0.f, 0.f, 0.f};
#pragma unroll
    for (int cc = 0; cc < 8; ++cc) {
        const float pc = p_sh[csub * 8 + cc];
        v8h vv = __builtin_nontemporal_load((const v8h*)(vrow + (size_t)cc * DHEAD));
#pragma unroll
        for (int i = 0; i < 8; ++i) oacc[i] += pc * (float)vv[i];
    }
#pragma unroll
    for (int i = 0; i < 8; ++i) atomicAdd(&o_sh[d0 + i], oacc[i]);
    __syncthreads();

    const size_t idx = (size_t)bh * NSPLITS + chunk;
    if (tid < DHEAD) o_ws[idx * DHEAD + tid] = o_sh[tid];
    if (tid == 0) { m_ws[idx] = mchunk; l_ws[idx] = lchunk; }
}

// ---------------------------------------------------------------------------
// Phase 2: one workgroup per (b*h). Cross-chunk online-softmax combine.
// ---------------------------------------------------------------------------
__global__ __launch_bounds__(128)
void fd_combine(const float* __restrict__ m_ws,
                const float* __restrict__ l_ws,
                const float* __restrict__ o_ws,
                _Float16* __restrict__ out)
{
    const int bh = blockIdx.x;
    const int t  = threadIdx.x;   // 128 threads

    __shared__ float red[128];
    __shared__ float alpha_sh[NSPLITS];

    const float* m = m_ws + (size_t)bh * NSPLITS;
    const float* l = l_ws + (size_t)bh * NSPLITS;

    // global max over 64 chunk maxima
    red[t] = (t < NSPLITS) ? m[t] : -1e30f;
    __syncthreads();
    for (int s2 = 64; s2 > 0; s2 >>= 1) {
        if (t < s2) red[t] = fmaxf(red[t], red[t + s2]);
        __syncthreads();
    }
    const float M = red[0];
    __syncthreads();

    // alpha and L = sum(l * alpha)
    float la = 0.f;
    if (t < NSPLITS) {
        const float a = __expf(m[t] - M);
        alpha_sh[t] = a;
        la = l[t] * a;
    }
    red[t] = la;
    __syncthreads();
    for (int s2 = 64; s2 > 0; s2 >>= 1) {
        if (t < s2) red[t] += red[t + s2];
        __syncthreads();
    }
    const float L = red[0];

    // O[d] = sum_n alpha[n] * o[n][d] / L
    const float* ob = o_ws + (size_t)bh * NSPLITS * DHEAD;
    float acc = 0.f;
#pragma unroll 8
    for (int n = 0; n < NSPLITS; ++n)
        acc += alpha_sh[n] * ob[(size_t)n * DHEAD + t];

    out[(size_t)bh * DHEAD + t] = (_Float16)(acc / L);
}

// ---------------------------------------------------------------------------
extern "C" void kernel_launch(void* const* d_in, const int* in_sizes, int n_in,
                              void* d_out, int out_size, void* d_ws, size_t ws_size,
                              hipStream_t stream)
{
    const _Float16* q = (const _Float16*)d_in[0];
    const _Float16* k = (const _Float16*)d_in[1];
    const _Float16* v = (const _Float16*)d_in[2];
    _Float16* out = (_Float16*)d_out;

    const int BH = in_sizes[0] / DHEAD;             // B*H (q is [B,H,1,D])
    const int S  = in_sizes[1] / in_sizes[0];       // 8192 ; chunk = S/64 = 128

    float* m_ws = (float*)d_ws;
    float* l_ws = m_ws + (size_t)BH * NSPLITS;
    float* o_ws = l_ws + (size_t)BH * NSPLITS;

    fd_partial<<<BH * NSPLITS, 256, 0, stream>>>(q, k, v, m_ws, l_ws, o_ws, S);
    fd_combine<<<BH, 128, 0, stream>>>(m_ws, l_ws, o_ws, out);
}
